// AxialAttention_52166672778072
// MI455X (gfx1250) — matrix operand order repaired and checked by reading, not compiled, not run on hardware
//
#include <hip/hip_runtime.h>

// ---------------------------------------------------------------------------
// AxialAttention for MI455X (gfx1250, wave32, WMMA 16x16x32 f16 + TDM)
// B=2, H=128, W=256, C=256, heads=8, d=32
// ---------------------------------------------------------------------------

typedef _Float16 h16 __attribute__((ext_vector_type(16)));
typedef _Float16 h2  __attribute__((ext_vector_type(2)));
typedef float    f8  __attribute__((ext_vector_type(8)));
typedef unsigned int v4u __attribute__((ext_vector_type(4)));
typedef int          v8i __attribute__((ext_vector_type(8)));
typedef int          v4i __attribute__((ext_vector_type(4)));

#define CB    256         // channels
#define NHEAD 8
#define DH    32          // dim_head
#define BB    2
#define HH    128
#define WW    256
#define HWSZ  (HH*WW)     // 32768
#define MROWS (BB*HWSZ)   // 65536

#if defined(__has_builtin)
#if __has_builtin(__builtin_amdgcn_tensor_load_to_lds) && \
    __has_builtin(__builtin_amdgcn_s_wait_tensorcnt)
#define HAVE_TDM 1
#endif
#endif
#ifndef HAVE_TDM
#define HAVE_TDM 0
#endif

__device__ __forceinline__ f8 wmma16(h16 a, h16 b, f8 c) {
    // D(16x16 f32) = A(16x32 f16) x B(32x16 f16) + C
    return __builtin_amdgcn_wmma_f32_16x16x32_f16(false, a, false, b,
                                                  (short)0, c, false, false);
}

// K index pattern for the 16-bit A-matrix fragment (contiguous order):
// VGPR vg holds halves (2vg, 2vg+1); lanes 0-15 -> K 0-15, lanes 16-31 -> K 16-31
__device__ __forceinline__ int a_kpat(int vg, int hi) {
    return hi * 16 + vg * 2;
}

// Load A fragment (16 rows x 32 k) from a row-major f16 array, row stride ldm.
__device__ __forceinline__ h16 load_a(const _Float16* base, int ldm, int lane) {
    int m = lane & 15, hi = lane >> 4;
    const _Float16* r = base + m * ldm;
    h16 a;
#pragma unroll
    for (int vg = 0; vg < 8; ++vg) {
        int k = a_kpat(vg, hi);
        h2 p = *(const h2*)(r + k);
        a[2 * vg]     = p.x;
        a[2 * vg + 1] = p.y;
    }
    return a;
}

#if HAVE_TDM
// TDM: async DMA of a 2D f16 tile (tile_d0 elems/row, tile_d1 rows,
// row stride in elements) from global memory to contiguous LDS.
// D# layout per CDNA5 ISA 8.3/8.4 (group0: count/lds/global/type,
// group1: data_size + tensor/tile dims + dim0 stride).
// This toolchain exposes the 6-arg builtin:
//   (uint32x4 g0, int32x8 g1, int32x4, int32x4, int32x8, i32 cpol)
__device__ __forceinline__ void tdm_load_2d(unsigned int lds_off,
                                            const void* gptr,
                                            unsigned int tile_d0,
                                            unsigned int tile_d1,
                                            unsigned long long stride_elems)
{
    unsigned long long ga = (unsigned long long)gptr;
    v4u g0;
    g0[0] = 1u;                                  // count=1, user descriptor
    g0[1] = lds_off;                             // lds_addr (bytes)
    g0[2] = (unsigned int)ga;                    // global_addr[31:0]
    g0[3] = (unsigned int)((ga >> 32) & 0x1FFFFFFu) | 0x80000000u; // hi + type=2
    v8i g1;
    g1[0] = 0x00010000;                          // data_size=1 (2B), mask=0
    g1[1] = (int)((tile_d0 & 0xFFFFu) << 16);    // tensor_dim0[15:0]
    g1[2] = (int)(((tile_d0 >> 16) & 0xFFFFu) |  // tensor_dim0[31:16]
                  ((tile_d1 & 0xFFFFu) << 16));  // tensor_dim1[15:0]
    g1[3] = (int)(((tile_d1 >> 16) & 0xFFFFu) |  // tensor_dim1[31:16]
                  ((tile_d0 & 0xFFFFu) << 16));  // tile_dim0
    g1[4] = (int)(tile_d1 & 0xFFFFu);            // tile_dim1 (tile_dim2=0)
    g1[5] = (int)(unsigned int)stride_elems;     // tensor_dim0_stride[31:0]
    g1[6] = (int)(unsigned int)((stride_elems >> 32) & 0xFFFFu); // stride[47:32]
    g1[7] = 0;                                   // tensor_dim1_stride hi (unused)
    v4i z4 = (v4i)0;
    v8i z8 = (v8i)0;
    __builtin_amdgcn_tensor_load_to_lds(g0, g1, z4, z4, z8, 0);
}
#endif

// ---------------------------------------------------------------------------
// GEMM: D[f16] = X[f32] * W[f32 as f16]   (M x 256 x 256)
// One 256-thread block computes a 64x256 tile. blockIdx.z picks Wq/Wk/Wv.
// W staged transposed in LDS so B fragments are contiguous 32B reads.
// ---------------------------------------------------------------------------
__global__ void gemm_qkv_kernel(const float* __restrict__ X,
                                const float* __restrict__ Wq,
                                const float* __restrict__ Wk,
                                const float* __restrict__ Wv,
                                _Float16* __restrict__ Dq,
                                _Float16* __restrict__ Dk,
                                _Float16* __restrict__ Dv)
{
    __shared__ __align__(32) _Float16 Xs[64 * 64];    // [m][k]
    __shared__ __align__(32) _Float16 Wts[256 * 64];  // [n][k] (transposed)

    const float* Wm = (blockIdx.z == 0) ? Wq : (blockIdx.z == 1) ? Wk : Wv;
    _Float16*    Dm = (blockIdx.z == 0) ? Dq : (blockIdx.z == 1) ? Dk : Dv;

    int m0   = blockIdx.x * 64;
    int tid  = threadIdx.x;
    int lane = tid & 31, wave = tid >> 5;
    int wr = wave >> 1, wc = wave & 1;   // 4 row-waves x 2 col-waves

    f8 acc[8];
#pragma unroll
    for (int i = 0; i < 8; ++i) acc[i] = (f8)0.0f;

    for (int kc = 0; kc < 4; ++kc) {     // K chunks of 64
        __syncthreads();
        if (kc < 3) {                    // prefetch next chunk into L2/L0
            __builtin_prefetch(&X[(size_t)(m0 + (tid >> 2)) * CB +
                                  (kc + 1) * 64 + (tid & 3) * 16], 0, 1);
            __builtin_prefetch(&Wm[(size_t)((kc + 1) * 64 + (tid >> 2)) * CB +
                                   (tid & 3) * 64], 0, 1);
        }
        for (int e = tid; e < 64 * 64; e += 256) {
            int m = e >> 6, k = e & 63;
            Xs[e] = (_Float16)X[(size_t)(m0 + m) * CB + kc * 64 + k];
        }
        for (int e = tid; e < 64 * 256; e += 256) {
            int kl = e >> 8, n = e & 255;
            Wts[n * 64 + kl] = (_Float16)Wm[(size_t)(kc * 64 + kl) * CB + n];
        }
        __syncthreads();
#pragma unroll
        for (int ks = 0; ks < 2; ++ks) {
            h16 a = load_a(Xs + wr * 16 * 64 + ks * 32, 64, lane);
            h16 bfr[8];
#pragma unroll
            for (int nt = 0; nt < 8; ++nt) {   // batch the B-frag loads
                int n = wc * 128 + nt * 16 + (lane & 15);
                bfr[nt] = *(const h16*)(Wts + n * 64 + ks * 32 + (lane >> 4) * 16);
            }
#pragma unroll
            for (int nt = 0; nt < 8; ++nt)
                acc[nt] = wmma16(a, bfr[nt], acc[nt]);
        }
    }
    int hi = lane >> 4, n0 = lane & 15;
#pragma unroll
    for (int nt = 0; nt < 8; ++nt) {
        int col = wc * 128 + nt * 16 + n0;
#pragma unroll
        for (int i = 0; i < 8; ++i) {
            int row = m0 + wr * 16 + i + hi * 8;
            Dm[(size_t)row * CB + col] = (_Float16)acc[nt][i];
        }
    }
}

// ---------------------------------------------------------------------------
// Output projection: out[f32] (=/+=) O[f16] * Wo + bo
// A-chunk staged by the Tensor Data Mover (f16 tile, no conversion needed).
// ---------------------------------------------------------------------------
__global__ void gemm_proj_kernel(const _Float16* __restrict__ A,
                                 const float* __restrict__ Wo,
                                 const float* __restrict__ bo,
                                 float* __restrict__ out,
                                 int accum)
{
    __shared__ __align__(32) _Float16 Xs[64 * 64];
    __shared__ __align__(32) _Float16 Wts[256 * 64];

    int m0   = blockIdx.x * 64;
    int tid  = threadIdx.x;
    int lane = tid & 31, wave = tid >> 5;
    int wr = wave >> 1, wc = wave & 1;

    f8 acc[8];
#pragma unroll
    for (int i = 0; i < 8; ++i) acc[i] = (f8)0.0f;

    for (int kc = 0; kc < 4; ++kc) {
        __syncthreads();
#if HAVE_TDM
        if (wave == 0) {
            tdm_load_2d((unsigned int)(unsigned long long)(uintptr_t)Xs,
                        A + (size_t)m0 * CB + kc * 64, 64, 64, CB);
            __builtin_amdgcn_s_wait_tensorcnt(0);
        }
#else
        for (int e = tid; e < 64 * 64; e += 256) {
            int m = e >> 6, k = e & 63;
            Xs[e] = A[(size_t)(m0 + m) * CB + kc * 64 + k];
        }
#endif
        for (int e = tid; e < 64 * 256; e += 256) {
            int kl = e >> 8, n = e & 255;
            Wts[n * 64 + kl] = (_Float16)Wo[(size_t)(kc * 64 + kl) * CB + n];
        }
        __syncthreads();
#pragma unroll
        for (int ks = 0; ks < 2; ++ks) {
            h16 a = load_a(Xs + wr * 16 * 64 + ks * 32, 64, lane);
            h16 bfr[8];
#pragma unroll
            for (int nt = 0; nt < 8; ++nt) {
                int n = wc * 128 + nt * 16 + (lane & 15);
                bfr[nt] = *(const h16*)(Wts + n * 64 + ks * 32 + (lane >> 4) * 16);
            }
#pragma unroll
            for (int nt = 0; nt < 8; ++nt)
                acc[nt] = wmma16(a, bfr[nt], acc[nt]);
        }
    }
    int hi = lane >> 4, n0 = lane & 15;
#pragma unroll
    for (int nt = 0; nt < 8; ++nt) {
        int col = wc * 128 + nt * 16 + n0;
        float bv = bo[col];
#pragma unroll
        for (int i = 0; i < 8; ++i) {
            int row = m0 + wr * 16 + i + hi * 8;
            size_t idx = (size_t)row * CB + col;
            float v = acc[nt][i] + bv;
            out[idx] = accum ? (out[idx] + v) : v;
        }
    }
}

// ---------------------------------------------------------------------------
// Flash axial attention. One wave32 workgroup = one (batch, slab, head).
// Row l of the slab lives at global row (base + l*stride).
//   height axis: Pmod=W, baseMul=1,  stride=W, L=H
//   width  axis: Pmod=H, baseMul=W,  stride=1, L=W
// K slab staged by the TDM (strided 2D f16 tile -> contiguous LDS).
// ---------------------------------------------------------------------------
__global__ void attn_kernel(const _Float16* __restrict__ Q,
                            const _Float16* __restrict__ K,
                            const _Float16* __restrict__ V,
                            _Float16* __restrict__ O,
                            int L, int Pmod, int baseMul, int stride)
{
    __shared__ __align__(32) _Float16 Ks[256 * 32];  // [l][d]  row-major
    __shared__ __align__(32) _Float16 Vt[32 * 256];  // [d][l]  transposed
    __shared__ __align__(32) _Float16 Pl[16 * 32];   // P bounce tile

    int head = blockIdx.y;
    int s = blockIdx.x;
    int b = s / Pmod, p = s % Pmod;
    size_t base = (size_t)b * HWSZ + (size_t)p * baseMul;

    int lane = threadIdx.x;               // 32 threads = 1 wave
    int n0 = lane & 15, hi = lane >> 4;
    const float scale = 0.17677669529663687f;   // 32^-0.5

    // Stage K rows: [l][32] contiguous in LDS
#if HAVE_TDM
    tdm_load_2d((unsigned int)(unsigned long long)(uintptr_t)Ks,
                K + base * CB + head * DH, DH, (unsigned int)L,
                (unsigned long long)stride * CB);
#else
    for (int e = lane; e < L * 4; e += 32) {
        int l = e >> 2, q = e & 3;
        const uint4* src = (const uint4*)(K + (base + (size_t)l * stride) * CB + head * DH);
        ((uint4*)Ks)[l * 4 + q] = src[q];
    }
#endif
    // Stage V transposed: read 16B of a row, scatter 8 halves into columns
    for (int e = lane; e < L * 4; e += 32) {
        int l = e >> 2, q = e & 3;
        const uint4* src = (const uint4*)(V + (base + (size_t)l * stride) * CB + head * DH);
        uint4 t = src[q];
        _Float16 tmp[8];
        *(uint4*)tmp = t;
#pragma unroll
        for (int j = 0; j < 8; ++j)
            Vt[(q * 8 + j) * L + l] = tmp[j];
    }
#if HAVE_TDM
    __builtin_amdgcn_s_wait_tensorcnt(0);
#endif
    __syncthreads();

    for (int qb = 0; qb < L / 16; ++qb) {
        // Q fragment straight from global (L2-resident)
        int mrow = qb * 16 + n0;
        const _Float16* qp = Q + (base + (size_t)mrow * stride) * CB + head * DH;
        h16 qa;
#pragma unroll
        for (int vg = 0; vg < 8; ++vg) {
            int k = a_kpat(vg, hi);
            h2 pr = *(const h2*)(qp + k);
            qa[2 * vg]     = pr.x;
            qa[2 * vg + 1] = pr.y;
        }

        f8 o0 = (f8)0.0f, o1 = (f8)0.0f;
        float mx[8], sm[8];
#pragma unroll
        for (int i = 0; i < 8; ++i) { mx[i] = -1e30f; sm[i] = 0.0f; }

        for (int jj = 0; jj < L; jj += 32) {
            // S = Q x K^T : B[k=d][n=kv] = Ks[jj+n][k], contiguous in k
            h16 kb0 = *(const h16*)(Ks + (jj + n0) * DH + hi * 16);
            h16 kb1 = *(const h16*)(Ks + (jj + 16 + n0) * DH + hi * 16);
            f8 z = (f8)0.0f;
            f8 s0 = wmma16(qa, kb0, z);
            f8 s1 = wmma16(qa, kb1, z);

            // online softmax; lane group (hi) holds rows i + hi*8, cols = n0
#pragma unroll
            for (int i = 0; i < 8; ++i) {
                float a = s0[i] * scale, c = s1[i] * scale;
                float cm = fmaxf(a, c);
#pragma unroll
                for (int msk = 1; msk < 16; msk <<= 1)
                    cm = fmaxf(cm, __shfl_xor(cm, msk, 32));
                float nmx = fmaxf(mx[i], cm);
                float al  = __expf(mx[i] - nmx);
                float p0  = __expf(a - nmx);
                float p1  = __expf(c - nmx);
                float r = p0 + p1;
#pragma unroll
                for (int msk = 1; msk < 16; msk <<= 1)
                    r += __shfl_xor(r, msk, 32);
                sm[i] = sm[i] * al + r;
                mx[i] = nmx;
                o0[i] *= al;
                o1[i] *= al;
                Pl[(i + hi * 8) * 32 + n0]      = (_Float16)p0;
                Pl[(i + hi * 8) * 32 + 16 + n0] = (_Float16)p1;
            }
            // reload P in A-fragment layout (same-wave LDS ordering is in-order)
            h16 pa;
            const _Float16* pr = Pl + n0 * 32;
#pragma unroll
            for (int vg = 0; vg < 8; ++vg) {
                int k = a_kpat(vg, hi);
                h2 t = *(const h2*)(pr + k);
                pa[2 * vg]     = t.x;
                pa[2 * vg + 1] = t.y;
            }
            // O += P x V : B[k=kv][n=d] = Vt[n][jj+k], contiguous in k
            h16 vb0 = *(const h16*)(Vt + n0 * L + jj + hi * 16);
            h16 vb1 = *(const h16*)(Vt + (16 + n0) * L + jj + hi * 16);
            o0 = wmma16(pa, vb0, o0);
            o1 = wmma16(pa, vb1, o1);
        }
#pragma unroll
        for (int i = 0; i < 8; ++i) {
            float inv = 1.0f / sm[i];
            int row = qb * 16 + i + hi * 8;
            _Float16* op = O + (base + (size_t)row * stride) * CB + head * DH;
            op[n0]      = (_Float16)(o0[i] * inv);
            op[16 + n0] = (_Float16)(o1[i] * inv);
        }
    }
}

// ---------------------------------------------------------------------------
extern "C" void kernel_launch(void* const* d_in, const int* in_sizes, int n_in,
                              void* d_out, int out_size, void* d_ws, size_t ws_size,
                              hipStream_t stream) {
    const float* x    = (const float*)d_in[0];
    const float* Wq_h = (const float*)d_in[1];
    const float* Wk_h = (const float*)d_in[2];
    const float* Wv_h = (const float*)d_in[3];
    const float* Wo_h = (const float*)d_in[4];
    const float* bo_h = (const float*)d_in[5];
    const float* Wq_w = (const float*)d_in[6];
    const float* Wk_w = (const float*)d_in[7];
    const float* Wv_w = (const float*)d_in[8];
    const float* Wo_w = (const float*)d_in[9];
    const float* bo_w = (const float*)d_in[10];
    float* out = (float*)d_out;

    char* ws = (char*)d_ws;
    const size_t seg = (size_t)MROWS * CB * sizeof(_Float16);  // 33.55 MB
    _Float16* Qb = (_Float16*)(ws);
    _Float16* Kb = (_Float16*)(ws + seg);
    _Float16* Vb = (_Float16*)(ws + 2 * seg);
    _Float16* Ob = (_Float16*)(ws + 3 * seg);

    dim3 gemmGrid(MROWS / 64, 1, 3);

    // ---- height-axis attention ----
    gemm_qkv_kernel<<<gemmGrid, 256, 0, stream>>>(x, Wq_h, Wk_h, Wv_h, Qb, Kb, Vb);
    attn_kernel<<<dim3(BB * WW, NHEAD), 32, 0, stream>>>(Qb, Kb, Vb, Ob,
                                                         HH, WW, 1, WW);
    gemm_proj_kernel<<<dim3(MROWS / 64), 256, 0, stream>>>(Ob, Wo_h, bo_h, out, 0);

    // ---- width-axis attention ----
    gemm_qkv_kernel<<<gemmGrid, 256, 0, stream>>>(x, Wq_w, Wk_w, Wv_w, Qb, Kb, Vb);
    attn_kernel<<<dim3(BB * HH, NHEAD), 32, 0, stream>>>(Qb, Kb, Vb, Ob,
                                                         WW, HH, WW, 1);
    gemm_proj_kernel<<<dim3(MROWS / 64), 256, 0, stream>>>(Ob, Wo_w, bo_w, out, 1);
}
